// Detector_81767587381901
// MI455X (gfx1250) — matrix operand order repaired and checked by reading, not compiled, not run on hardware
//
#include <hip/hip_runtime.h>

typedef _Float16 v16h __attribute__((ext_vector_type(16)));
typedef _Float16 v8h  __attribute__((ext_vector_type(8)));
typedef float    v8f  __attribute__((ext_vector_type(8)));

// ---------------------------------------------------------------------------
// TDM support — this toolchain has the 6-arg builtin:
//   (uint32x4 g0, int32x8 g1, int32x4 g2, int32x4 g3, int32x8 gx, i32 cpol)
// ---------------------------------------------------------------------------
#if defined(__has_builtin)
#if __has_builtin(__builtin_amdgcn_tensor_load_to_lds) && __has_builtin(__builtin_amdgcn_s_wait_tensorcnt)
#define HAVE_TDM 1
#endif
#endif
#ifndef HAVE_TDM
#define HAVE_TDM 0
#endif

#if defined(__gfx1250__) && HAVE_TDM
typedef unsigned int tdm_u32x4 __attribute__((ext_vector_type(4)));
typedef int          tdm_i32x4 __attribute__((ext_vector_type(4)));
typedef int          tdm_i32x8 __attribute__((ext_vector_type(8)));

// 2D tile load: tile_k x tile_n halves, global row stride = stride_k halves,
// remaining extents rem_k/rem_n for HW zero-fill of OOB reads.
__device__ __forceinline__ void tdm_load_2d(unsigned lds_off, const void* gaddr,
                                            unsigned rem_k, unsigned rem_n,
                                            unsigned stride_k,
                                            unsigned tile_k, unsigned tile_n)
{
    unsigned long long ga = (unsigned long long)(uintptr_t)gaddr;
    tdm_u32x4 g0;
    g0[0] = 1u;                                             // count=1, user mode
    g0[1] = lds_off;                                        // LDS byte address
    g0[2] = (unsigned)(ga & 0xffffffffu);                   // global_addr[31:0]
    g0[3] = (unsigned)((ga >> 32) & 0x01ffffffu) | (2u << 30); // addr[56:32] | type=2
    tdm_i32x8 g1;
    g1[0] = (int)(1u << 16);                                // data_size=1 (2 bytes)
    g1[1] = (int)((rem_k & 0xffffu) << 16);                 // tensor_dim0[15:0]
    g1[2] = (int)(((rem_k >> 16) & 0xffffu) | ((rem_n & 0xffffu) << 16));
    g1[3] = (int)(((rem_n >> 16) & 0xffffu) | ((tile_k & 0xffffu) << 16)); // tile_dim0
    g1[4] = (int)(tile_n & 0xffffu);                        // tile_dim1 (tile_dim2=0)
    g1[5] = (int)stride_k;                                  // tensor_dim0_stride[31:0]
    g1[6] = 0;
    g1[7] = 0;
    tdm_i32x4 z4 = {0, 0, 0, 0};
    tdm_i32x8 z8 = {0, 0, 0, 0, 0, 0, 0, 0};
    __builtin_amdgcn_tensor_load_to_lds(g0, g1, z4, z4, z8, 0);
}
#endif

// ---------------------------------------------------------------------------
// Fragment loaders (LDS row stride = 64 halves in all kernels this round)
// A frag (16x32 MxK): lane<16 -> K ko+[0..7 , 16..23]; lane>=16 -> ko+[8..15, 24..31]
// B frag (32x16 KxN) from sB[n][k]: lane<16 -> K ko+[0..7 , 16..23]; lane>=16 -> ko+[8..15, 24..31]
// ---------------------------------------------------------------------------
__device__ __forceinline__ v16h load_afrag(const _Float16* sA, int row0, int ko, int lane) {
    int row  = row0 + (lane & 15);
    int base = ko + ((lane < 16) ? 0 : 8);
    const _Float16* p = &sA[row * 64 + base];
    v8h lo = *(const v8h*)p;
    v8h hi = *(const v8h*)(p + 16);
    return __builtin_shufflevector(lo, hi, 0,1,2,3,4,5,6,7,8,9,10,11,12,13,14,15);
}
__device__ __forceinline__ v16h load_bfrag(const _Float16* sB, int col0, int ko, int lane) {
    int col  = col0 + (lane & 15);
    int base = ko + ((lane < 16) ? 0 : 8);
    const _Float16* p = &sB[col * 64 + base];
    v8h lo = *(const v8h*)p;
    v8h hi = *(const v8h*)(p + 16);
    return __builtin_shufflevector(lo, hi, 0,1,2,3,4,5,6,7,8,9,10,11,12,13,14,15);
}

// ---------------------------------------------------------------------------
// f32 -> f16 elementwise (image)
// ---------------------------------------------------------------------------
__global__ void cvt_f32_f16(const float* __restrict__ src, _Float16* __restrict__ dst, int n) {
    int i = blockIdx.x * blockDim.x + threadIdx.x;
    if (i < n) dst[i] = (_Float16)src[i];
}

// f32 HWIO -> f16 transposed [Cout][Ktot]  (dst[co*Ktot + k] = src[k*Cout + co])
__global__ void cvt_wgt_t(const float* __restrict__ src, _Float16* __restrict__ dst,
                          int Ktot, int Cout) {
    int i = blockIdx.x * blockDim.x + threadIdx.x;
    if (i >= Ktot * Cout) return;
    int co = i / Ktot;
    int k  = i - co * Ktot;
    dst[i] = (_Float16)src[k * Cout + co];
}

// ---------------------------------------------------------------------------
// Implicit-GEMM 3x3 SAME conv + bias + relu. f16 in/out, f32 accumulate.
// Macro tile 128(M) x 64(N), K-stage 64. 8 waves as 4(M) x 2(N), each wave
// owns 32x32 = 2x2 WMMA tiles -> 8 WMMA per wave per stage.
// grid = (H*W/128, Cout/64). Weights pre-transposed to [Cout][9*Cin] f16.
// ---------------------------------------------------------------------------
template <int CIN>
__global__ __launch_bounds__(256)
void conv_wmma(const _Float16* __restrict__ in, const _Float16* __restrict__ wgtT,
               const float* __restrict__ bias, _Float16* __restrict__ out,
               int lw, int Cout)
{
    constexpr int KTOT = 9 * CIN;
    const int W = 1 << lw;
    const int H = W;

    __shared__ __align__(16) _Float16 sA[128 * 64];   // [m][k]
    __shared__ __align__(16) _Float16 sB[64 * 64];    // [n][k]

    const int M0   = blockIdx.x * 128;
    const int N0   = blockIdx.y * 64;
    const int tid  = threadIdx.x;
    const int lane = tid & 31;
    const int wave = tid >> 5;
    const int mi   = wave >> 1;   // 0..3
    const int ni   = wave & 1;    // 0..1

    v8f acc[2][2] = {};

    // Per-thread K decomposition for the vectorized A gather (invariant part)
    const int kk8 = (tid & 7) * 8;     // 0..56, this thread's K sub-offset
    const int mrb = tid >> 3;          // 0..31, this thread's base M row

    for (int k0 = 0; k0 < KTOT; k0 += 64) {
        // ---------------- stage B tile via TDM (wave 0) ----------------
#if defined(__gfx1250__) && HAVE_TDM
        if (wave == 0) {
            unsigned lds_off = (unsigned)(uintptr_t)(void*)&sB[0];
            tdm_load_2d(lds_off, wgtT + (size_t)N0 * KTOT + k0,
                        (unsigned)(KTOT - k0), (unsigned)(Cout - N0),
                        (unsigned)KTOT, 64u, 64u);
        }
#else
        for (int e = tid; e < 64 * 64; e += 256) {
            int n  = e >> 6;
            int kk = e & 63;
            int k  = k0 + kk;
            _Float16 v = (_Float16)0.0f;
            if (k < KTOT) v = wgtT[(size_t)(N0 + n) * KTOT + k];
            sB[n * 64 + kk] = v;
        }
#endif

        // ---------------- stage A tile (gather with SAME padding) ------
        if constexpr (CIN % 8 == 0) {
            // K is channel-contiguous within one filter tap: b128 path.
            const int k  = k0 + kk8;
            const int t  = k / CIN;              // constexpr divisor
            const int ci = k - t * CIN;
            const int ky = t / 3, kx = t - ky * 3;
            #pragma unroll
            for (int j = 0; j < 4; ++j) {
                int mr = mrb + j * 32;
                int m  = M0 + mr;
                int y  = m >> lw, x = m & (W - 1);
                int iy = y + ky - 1, ix = x + kx - 1;
                v8h v = {};
                if (iy >= 0 && iy < H && ix >= 0 && ix < W)
                    v = *(const v8h*)&in[(size_t)((iy << lw) + ix) * CIN + ci];
                *(v8h*)&sA[mr * 64 + kk8] = v;
            }
        } else {
            // CIN==3 (layer 0): scalar masked path, single K stage (KTOT=27)
            for (int e = tid; e < 128 * 64; e += 256) {
                int mr = e >> 6;
                int kk = e & 63;
                int k  = k0 + kk;
                _Float16 v = (_Float16)0.0f;
                if (k < KTOT) {
                    int ci = k % CIN;
                    int t  = k / CIN;
                    int ky = t / 3, kx = t - ky * 3;
                    int m  = M0 + mr;
                    int y  = m >> lw, x = m & (W - 1);
                    int iy = y + ky - 1, ix = x + kx - 1;
                    if (iy >= 0 && iy < H && ix >= 0 && ix < W)
                        v = in[(size_t)((iy << lw) + ix) * CIN + ci];
                }
                sA[mr * 64 + kk] = v;
            }
        }

#if defined(__gfx1250__) && HAVE_TDM
        __builtin_amdgcn_s_wait_tensorcnt(0);
#endif
        __syncthreads();

        // ---------------- compute: 2 K-steps x 2x2 tiles ----------------
        #pragma unroll
        for (int ks = 0; ks < 2; ++ks) {
            v16h af[2], bf[2];
            #pragma unroll
            for (int mt = 0; mt < 2; ++mt)
                af[mt] = load_afrag(sA, mi * 32 + mt * 16, ks * 32, lane);
            #pragma unroll
            for (int nt = 0; nt < 2; ++nt)
                bf[nt] = load_bfrag(sB, ni * 32 + nt * 16, ks * 32, lane);
            #pragma unroll
            for (int mt = 0; mt < 2; ++mt)
                #pragma unroll
                for (int nt = 0; nt < 2; ++nt)
                    acc[mt][nt] = __builtin_amdgcn_wmma_f32_16x16x32_f16(
                        false, af[mt], false, bf[nt], (short)0, acc[mt][nt],
                        false, false);
        }
        __syncthreads();
    }

    // ---------------- epilogue: bias + relu, f16 store ------------------
    #pragma unroll
    for (int nt = 0; nt < 2; ++nt) {
        int n   = N0 + ni * 32 + nt * 16 + (lane & 15);
        float bv = bias[n];
        #pragma unroll
        for (int mt = 0; mt < 2; ++mt) {
            int m0 = M0 + mi * 32 + mt * 16 + ((lane < 16) ? 0 : 8);
            #pragma unroll
            for (int r = 0; r < 8; ++r) {
                float v = acc[mt][nt][r] + bv;
                v = v > 0.0f ? v : 0.0f;
                out[(size_t)(m0 + r) * Cout + n] = (_Float16)v;
            }
        }
    }
}

// ---------------------------------------------------------------------------
// 2x2 stride-2 maxpool, f16
// ---------------------------------------------------------------------------
__global__ void maxpool2(const _Float16* __restrict__ in, _Float16* __restrict__ out,
                         int H, int W, int C)
{
    int Ho = H >> 1, Wo = W >> 1;
    int total = Ho * Wo * C;
    int i = blockIdx.x * blockDim.x + threadIdx.x;
    if (i >= total) return;
    int c = i % C;
    int x = (i / C) % Wo;
    int y = i / (C * Wo);
    int base = ((2 * y) * W + 2 * x) * C + c;
    float a = (float)in[base];
    float b = (float)in[base + C];
    float d = (float)in[base + W * C];
    float e = (float)in[base + W * C + C];
    float m  = a > b ? a : b;
    float n2 = d > e ? d : e;
    out[i] = (_Float16)(m > n2 ? m : n2);
}

// ---------------------------------------------------------------------------
// ROI pooling (bilinear, 7x7), feat 32x32x512 f16 -> pooled f32 [64][7][7][512]
// ---------------------------------------------------------------------------
__global__ void roi_pool(const _Float16* __restrict__ feat, const float* __restrict__ rois,
                         float* __restrict__ pooled)
{
    const int H = 32, W = 32, C = 512;
    int roi = blockIdx.x;
    float cx = rois[roi * 4 + 0], cy = rois[roi * 4 + 1];
    float w  = rois[roi * 4 + 2], h  = rois[roi * 4 + 3];
    float r0 = roundf((cx - w * 0.5f) / 16.0f);   // x base
    float c0 = roundf((cy - h * 0.5f) / 16.0f);   // y base
    float wp = roundf(w / 16.0f);
    float hp = roundf(h / 16.0f);

    for (int e = threadIdx.x; e < 7 * 7 * 512; e += blockDim.x) {
        int c  = e % 512;
        int px = (e / 512) % 7;
        int py = e / (512 * 7);
        float gy = (float)py + 0.5f;
        float gx = (float)px + 0.5f;
        float hm1 = hp - 1.0f; hm1 = hm1 > 0.0f ? hm1 : 0.0f;
        float wm1 = wp - 1.0f; wm1 = wm1 > 0.0f ? wm1 : 0.0f;
        float ys = gy * hp / 7.0f - 0.5f;
        ys = ys < 0.0f ? 0.0f : (ys > hm1 ? hm1 : ys);
        ys += c0;
        float xs = gx * wp / 7.0f - 0.5f;
        xs = xs < 0.0f ? 0.0f : (xs > wm1 ? wm1 : xs);
        xs += r0;
        ys = ys < 0.0f ? 0.0f : (ys > (float)(H - 1) ? (float)(H - 1) : ys);
        xs = xs < 0.0f ? 0.0f : (xs > (float)(W - 1) ? (float)(W - 1) : xs);
        float y0f = floorf(ys), x0f = floorf(xs);
        int y0 = (int)y0f, x0 = (int)x0f;
        int y1 = y0 + 1 < H - 1 ? y0 + 1 : H - 1;
        int x1 = x0 + 1 < W - 1 ? x0 + 1 : W - 1;
        float wy = ys - y0f, wx = xs - x0f;
        float f00 = (float)feat[(y0 * W + x0) * C + c];
        float f01 = (float)feat[(y0 * W + x1) * C + c];
        float f10 = (float)feat[(y1 * W + x0) * C + c];
        float f11 = (float)feat[(y1 * W + x1) * C + c];
        float top = f00 * (1.0f - wx) + f01 * wx;
        float bot = f10 * (1.0f - wx) + f11 * wx;
        pooled[roi * 25088 + e] = top * (1.0f - wy) + bot * wy;
    }
}

// ---------------------------------------------------------------------------
// FC GEMM with WMMA: M=64, N=4096, K=25088. fp32 A/B converted to f16 while
// staging. Macro tile 64(M) x 128(N), K-stage 64; 8 waves as 2(M) x 4(N),
// each wave 32x32 = 2x2 tiles -> 8 WMMA per wave per stage. grid.x = N/128.
// ---------------------------------------------------------------------------
__global__ __launch_bounds__(256)
void fc_wmma(const float* __restrict__ A, const float* __restrict__ Bw,
             const float* __restrict__ bias, float* __restrict__ out,
             int N, int K)
{
    __shared__ __align__(16) _Float16 sA[64 * 64];    // [m][k]
    __shared__ __align__(16) _Float16 sB[128 * 64];   // [n][k]

    const int N0   = blockIdx.x * 128;
    const int tid  = threadIdx.x;
    const int lane = tid & 31;
    const int wave = tid >> 5;
    const int mi   = wave >> 2;   // 0..1
    const int ni   = wave & 3;    // 0..3

    v8f acc[2][2] = {};

    for (int k0 = 0; k0 < K; k0 += 64) {
        // A: 64x64, coalesced along K
        for (int e = tid; e < 64 * 64; e += 256) {
            int mr = e >> 6;
            int kk = e & 63;
            sA[mr * 64 + kk] = (_Float16)A[(size_t)mr * K + k0 + kk];
        }
        // B: sB[n][k] = Bw[(k0+kk)*N + N0+n], coalesced along n
        for (int e = tid; e < 128 * 64; e += 256) {
            int kk = e >> 7;
            int n  = e & 127;
            sB[n * 64 + kk] = (_Float16)Bw[(size_t)(k0 + kk) * N + (N0 + n)];
        }
        __syncthreads();

        #pragma unroll
        for (int ks = 0; ks < 2; ++ks) {
            v16h af[2], bf[2];
            #pragma unroll
            for (int mt = 0; mt < 2; ++mt)
                af[mt] = load_afrag(sA, mi * 32 + mt * 16, ks * 32, lane);
            #pragma unroll
            for (int nt = 0; nt < 2; ++nt)
                bf[nt] = load_bfrag(sB, ni * 32 + nt * 16, ks * 32, lane);
            #pragma unroll
            for (int mt = 0; mt < 2; ++mt)
                #pragma unroll
                for (int nt = 0; nt < 2; ++nt)
                    acc[mt][nt] = __builtin_amdgcn_wmma_f32_16x16x32_f16(
                        false, af[mt], false, bf[nt], (short)0, acc[mt][nt],
                        false, false);
        }
        __syncthreads();
    }

    #pragma unroll
    for (int nt = 0; nt < 2; ++nt) {
        int n   = N0 + ni * 32 + nt * 16 + (lane & 15);
        float bv = bias[n];
        #pragma unroll
        for (int mt = 0; mt < 2; ++mt) {
            int m0 = mi * 32 + mt * 16 + ((lane < 16) ? 0 : 8);
            #pragma unroll
            for (int r = 0; r < 8; ++r) {
                float v = acc[mt][nt][r] + bv;
                v = v > 0.0f ? v : 0.0f;
                out[(size_t)(m0 + r) * N + n] = v;
            }
        }
    }
}

// ---------------------------------------------------------------------------
// Heads: cls (4096->21 + softmax) and reg (4096->84). One block per ROI.
// out layout: [64*21 cls probs][64*84 reg] (tuple concat, flat).
// ---------------------------------------------------------------------------
__global__ void heads(const float* __restrict__ fc,
                      const float* __restrict__ cls_w, const float* __restrict__ cls_b,
                      const float* __restrict__ reg_w, const float* __restrict__ reg_b,
                      float* __restrict__ out)
{
    __shared__ float scls[21];
    int roi = blockIdx.x;
    int t = threadIdx.x;
    const float* f = fc + roi * 4096;

    if (t < 21) {
        float s = cls_b[t];
        for (int k = 0; k < 4096; ++k) s += f[k] * cls_w[k * 21 + t];
        scls[t] = s;
    } else if (t < 105) {
        int n = t - 21;
        float s = reg_b[n];
        for (int k = 0; k < 4096; ++k) s += f[k] * reg_w[k * 84 + n];
        out[64 * 21 + roi * 84 + n] = s;
    }
    __syncthreads();
    if (t < 21) {
        float mx = scls[0];
        for (int i = 1; i < 21; ++i) mx = scls[i] > mx ? scls[i] : mx;
        float sum = 0.0f;
        for (int i = 0; i < 21; ++i) sum += __expf(scls[i] - mx);
        out[roi * 21 + t] = __expf(scls[t] - mx) / sum;
    }
}

// ---------------------------------------------------------------------------
// Host launcher
// ---------------------------------------------------------------------------
extern "C" void kernel_launch(void* const* d_in, const int* in_sizes, int n_in,
                              void* d_out, int out_size, void* d_ws, size_t ws_size,
                              hipStream_t stream)
{
    const float* image = (const float*)d_in[0];
    const float* rois  = (const float*)d_in[1];
    const float* conv_w[13];
    const float* conv_b[13];
    for (int i = 0; i < 13; ++i) {
        conv_w[i] = (const float*)d_in[2 + i];
        conv_b[i] = (const float*)d_in[15 + i];
    }
    const float* fc_w  = (const float*)d_in[28];
    const float* fc_b  = (const float*)d_in[29];
    const float* cls_w = (const float*)d_in[30];
    const float* cls_b = (const float*)d_in[31];
    const float* reg_w = (const float*)d_in[32];
    const float* reg_b = (const float*)d_in[33];

    // workspace carve-up
    char* ws = (char*)d_ws;
    const size_t ACT_BYTES = (size_t)512 * 512 * 64 * sizeof(_Float16);  // 32 MB
    _Float16* actA  = (_Float16*)(ws);
    _Float16* actB  = (_Float16*)(ws + ACT_BYTES);
    _Float16* wgt16 = (_Float16*)(ws + 2 * ACT_BYTES);

    static const int ch[14] = {3, 64, 64, 128, 128, 256, 256, 256, 512, 512, 512, 512, 512, 512};
    size_t woff[13];
    size_t wtot = 0;
    for (int i = 0; i < 13; ++i) {
        woff[i] = wtot;
        wtot += (size_t)9 * ch[i] * ch[i + 1];
    }
    float* pooled = (float*)(ws + 2 * ACT_BYTES + ((wtot * sizeof(_Float16) + 255) & ~(size_t)255));
    float* fcbuf  = pooled + (size_t)64 * 25088;

    // convert conv weights (transposed to [Cout][Ktot]) + image to f16
    for (int i = 0; i < 13; ++i) {
        int Ktot = 9 * ch[i];
        int Cout = ch[i + 1];
        int n = Ktot * Cout;
        cvt_wgt_t<<<(n + 255) / 256, 256, 0, stream>>>(conv_w[i], wgt16 + woff[i], Ktot, Cout);
    }
    {
        int n = 512 * 512 * 3;
        cvt_f32_f16<<<(n + 255) / 256, 256, 0, stream>>>(image, actA, n);
    }

    // backbone
    _Float16* cur = actA;
    _Float16* nxt = actB;
    int lw = 9;   // W = 512
    int Cin = 3;
    for (int i = 0; i < 13; ++i) {
        int W = 1 << lw;
        int Cout = ch[i + 1];
        dim3 grid(W * W / 128, Cout / 64);
        switch (Cin) {
        case 3:   conv_wmma<3>  <<<grid, 256, 0, stream>>>(cur, wgt16 + woff[i], conv_b[i], nxt, lw, Cout); break;
        case 64:  conv_wmma<64> <<<grid, 256, 0, stream>>>(cur, wgt16 + woff[i], conv_b[i], nxt, lw, Cout); break;
        case 128: conv_wmma<128><<<grid, 256, 0, stream>>>(cur, wgt16 + woff[i], conv_b[i], nxt, lw, Cout); break;
        case 256: conv_wmma<256><<<grid, 256, 0, stream>>>(cur, wgt16 + woff[i], conv_b[i], nxt, lw, Cout); break;
        default:  conv_wmma<512><<<grid, 256, 0, stream>>>(cur, wgt16 + woff[i], conv_b[i], nxt, lw, Cout); break;
        }
        { _Float16* t = cur; cur = nxt; nxt = t; }
        Cin = Cout;
        if (i == 1 || i == 3 || i == 6 || i == 9) {
            int total = (W / 2) * (W / 2) * Cin;
            maxpool2<<<(total + 255) / 256, 256, 0, stream>>>(cur, nxt, W, W, Cin);
            { _Float16* t = cur; cur = nxt; nxt = t; }
            lw -= 1;
        }
    }
    // cur = feat 32x32x512 f16

    roi_pool<<<64, 256, 0, stream>>>(cur, rois, pooled);

    fc_wmma<<<4096 / 128, 256, 0, stream>>>(pooled, fc_w, fc_b, fcbuf, 4096, 25088);

    heads<<<64, 128, 0, stream>>>(fcbuf, cls_w, cls_b, reg_w, reg_b, (float*)d_out);
}
